// GraphDenoiseConvV60_37847251812402
// MI455X (gfx1250) — compile-verified, matrix operand order checked by reading, not run on hardware
//
#include <hip/hip_runtime.h>

// ---------------------------------------------------------------------------
// Types for CDNA5 WMMA (wave32, 16x16x32 bf16 -> f32)
// ---------------------------------------------------------------------------
typedef __attribute__((ext_vector_type(16))) __bf16 v16bf;
typedef __attribute__((ext_vector_type(8)))  float  v8f;

__device__ __forceinline__ unsigned short f2bf(float f) {
    unsigned u = __builtin_bit_cast(unsigned, f);
    unsigned r = u + 0x7FFFu + ((u >> 16) & 1u);   // round-to-nearest-even
    return (unsigned short)(r >> 16);
}
__device__ __forceinline__ float silu(float x) {
    return x / (1.f + __expf(-x));
}

// LDS byte offset = low 32 bits of the generic address (flat aperture rule).
__device__ __forceinline__ unsigned ldsaddr(const void* p) {
    return (unsigned)(unsigned long long)p;
}
// Async DMA: 16B per lane, global -> LDS, tracked by ASYNCcnt.
__device__ __forceinline__ void async_lds_b128(unsigned lds, unsigned goff,
                                               const void* base) {
    asm volatile("global_load_async_to_lds_b128 %0, %1, %2"
                 :: "v"(lds), "v"(goff), "s"((unsigned long long)base)
                 : "memory");
}
__device__ __forceinline__ void wait_async0() {
    asm volatile("s_wait_asynccnt 0x0" ::: "memory");
}

// ---------------------------------------------------------------------------
// Pack f32 weights (K x N, row-major) -> bf16 [N][Kpad] (transposed, padded)
// ---------------------------------------------------------------------------
__global__ void pack_w(const float* __restrict__ W, unsigned short* __restrict__ out,
                       int K, int N, int Kpad) {
    int idx = blockIdx.x * 256 + threadIdx.x;
    int total = N * Kpad;
    if (idx >= total) return;
    int n = idx / Kpad, k = idx - n * Kpad;
    float x = (k < K) ? W[(size_t)k * N + n] : 0.f;
    out[idx] = f2bf(x);
}

// ---------------------------------------------------------------------------
// Tiled WMMA GEMM:  out[M,N] = act( A[M,Kpad] @ Wp[N,Kpad]^T + bias )
// Workgroup tile 64x128, 8 waves (4m x 2n), wave tile 16x64, BK = 32.
// Double-buffered LDS; B tiles (and bf16 A tiles) staged by async DMA.
//   MODE 0: A = [v_f | p_pe | t_pe]  f32->bf16 (VALU)   Kpad 576
//   MODE 1: A = [e_f | t_pe]         f32->bf16 (VALU)   Kpad 640
//   MODE 2: A = [v[i] | v[j] | e]    bf16 async         Kpad 1024
//   MODE 3: A = h                    bf16 async         Kpad 512
//   MODE 4: A = v                    bf16 async         Kpad 256
//   MODE 5: A = [v2 | pooled[b]]     bf16 async         Kpad 1024
// ---------------------------------------------------------------------------
constexpr int G_BM = 64, G_BN = 128, G_BK = 32, G_LD = 40;
constexpr int AS_E = G_BM * G_LD;            // 2560
constexpr int BS_E = G_BN * G_LD;            // 5120
constexpr int BUF_E = AS_E + BS_E;           // 7680

template <int MODE>
__device__ __forceinline__ void stageA(
    unsigned short* __restrict__ As, int m0, int k0, int t,
    const float* __restrict__ f0, const float* __restrict__ f1,
    const float* __restrict__ f2,
    const unsigned short* __restrict__ b0, const unsigned short* __restrict__ b1) {
    const int arow = t >> 2;          // 0..63
    const int acb  = (t & 3) * 8;     // 0/8/16/24
    const int m = m0 + arow;
    const int c = k0 + acb;
    if (MODE == 0) {
        int b = m >> 7;
        unsigned pk[4];
        #pragma unroll
        for (int e2 = 0; e2 < 4; ++e2) {
            float x[2];
            #pragma unroll
            for (int h = 0; h < 2; ++h) {
                int cc = c + 2 * e2 + h;
                float v = 0.f;
                if (cc < 256)      v = f0[(size_t)m * 256 + cc];
                else if (cc < 456) v = f1[(size_t)m * 200 + (cc - 256)];
                else if (cc < 556) v = f2[b * 100 + (cc - 456)];
                x[h] = v;
            }
            pk[e2] = (unsigned)f2bf(x[0]) | ((unsigned)f2bf(x[1]) << 16);
        }
        *(uint4*)&As[arow * G_LD + acb] = make_uint4(pk[0], pk[1], pk[2], pk[3]);
    } else if (MODE == 1) {
        int b = m >> 14;
        uint4 w;
        if (c < 512) {  // chunk fully inside e_f (boundary 512 is a multiple of 8)
            const float4 x0 = *(const float4*)&f0[(size_t)m * 512 + c];
            const float4 x1 = *(const float4*)&f0[(size_t)m * 512 + c + 4];
            w.x = (unsigned)f2bf(x0.x) | ((unsigned)f2bf(x0.y) << 16);
            w.y = (unsigned)f2bf(x0.z) | ((unsigned)f2bf(x0.w) << 16);
            w.z = (unsigned)f2bf(x1.x) | ((unsigned)f2bf(x1.y) << 16);
            w.w = (unsigned)f2bf(x1.z) | ((unsigned)f2bf(x1.w) << 16);
        } else {        // t_pe tail + zero pad
            unsigned pk[4];
            #pragma unroll
            for (int e2 = 0; e2 < 4; ++e2) {
                int c0 = c + 2 * e2, c1 = c0 + 1;
                float xa = (c0 < 612) ? f2[b * 100 + c0 - 512] : 0.f;
                float xb = (c1 < 612) ? f2[b * 100 + c1 - 512] : 0.f;
                pk[e2] = (unsigned)f2bf(xa) | ((unsigned)f2bf(xb) << 16);
            }
            w = make_uint4(pk[0], pk[1], pk[2], pk[3]);
        }
        *(uint4*)&As[arow * G_LD + acb] = w;
    } else {
        unsigned lds = ldsaddr(&As[arow * G_LD + acb]);
        if (MODE == 2) {
            int b = m >> 14, p = m & 16383, i = p >> 7, j = p & 127;
            if (k0 < 512) {  // [v_i | v_j] region: same base (vb), per-lane row/col
                unsigned row = (c < 256) ? (unsigned)(b * 128 + i)
                                         : (unsigned)(b * 128 + j);
                unsigned cc  = (c < 256) ? (unsigned)c : (unsigned)(c - 256);
                async_lds_b128(lds, (row * 256u + cc) * 2u, b0);
            } else {         // e region
                async_lds_b128(lds, ((unsigned)m * 512u + (unsigned)(c - 512)) * 2u, b1);
            }
        } else if (MODE == 3) {
            async_lds_b128(lds, ((unsigned)m * 512u + (unsigned)c) * 2u, b0);
        } else if (MODE == 4) {
            async_lds_b128(lds, ((unsigned)m * 256u + (unsigned)c) * 2u, b0);
        } else { // MODE 5
            if (k0 < 512)
                async_lds_b128(lds, ((unsigned)m * 512u + (unsigned)c) * 2u, b0);
            else
                async_lds_b128(lds, ((unsigned)(m >> 7) * 512u + (unsigned)(c - 512)) * 2u, b1);
        }
    }
}

__device__ __forceinline__ void stageB(
    unsigned short* __restrict__ Bs, const unsigned short* __restrict__ Wp,
    int n0, int k0, int Kpad, int t) {
    const int brow = t >> 1;          // 0..127
    const int bcb  = (t & 1) * 16;    // 0/16
    unsigned goff = ((unsigned)(n0 + brow) * (unsigned)Kpad +
                     (unsigned)(k0 + bcb)) * 2u;
    unsigned lds = ldsaddr(&Bs[brow * G_LD + bcb]);
    async_lds_b128(lds,      goff,       Wp);
    async_lds_b128(lds + 16, goff + 16,  Wp);
}

template <int MODE>
__global__ __launch_bounds__(256) void gemm_wmma(
    const float* __restrict__ f0, const float* __restrict__ f1,
    const float* __restrict__ f2,
    const unsigned short* __restrict__ b0, const unsigned short* __restrict__ b1,
    const unsigned short* __restrict__ Wp, const float* __restrict__ bias,
    float* __restrict__ outF, unsigned short* __restrict__ outB,
    int M, int N, int Kpad, int act) {
    __shared__ unsigned short smem[2 * BUF_E];

    const int t = threadIdx.x;
    const int m0 = blockIdx.y * G_BM, n0 = blockIdx.x * G_BN;
    const int lane = t & 31, w = t >> 5;
    const int half = lane >> 4, r = lane & 15;
    const int wm = w >> 1, wn = w & 1;

    v8f acc[4] = {};
    const int nk = Kpad / G_BK;

    // prologue: stage tile 0
    stageA<MODE>(&smem[0], m0, 0, t, f0, f1, f2, b0, b1);
    stageB(&smem[AS_E], Wp, n0, 0, Kpad, t);
    wait_async0();
    __syncthreads();

    for (int kt = 0; kt < nk; ++kt) {
        unsigned short* As = &smem[(kt & 1) * BUF_E];
        unsigned short* Bs = As + AS_E;
        if (kt + 1 < nk) {  // issue DMA + staging for next tile (other buffer)
            unsigned short* As2 = &smem[((kt + 1) & 1) * BUF_E];
            stageA<MODE>(As2, m0, (kt + 1) * G_BK, t, f0, f1, f2, b0, b1);
            stageB(As2 + AS_E, Wp, n0, (kt + 1) * G_BK, Kpad, t);
        }
        // ---- A fragment (ISA 16-bit A 16x32 layout) ----
        v16bf af;
        const unsigned short* Ap = &As[(wm * 16 + r) * G_LD];
        #pragma unroll
        for (int e = 0; e < 8; ++e) {
            af[e]     = __builtin_bit_cast(__bf16, Ap[half * 8 + e]);
            af[e + 8] = __builtin_bit_cast(__bf16, Ap[16 + half * 8 + e]);
        }
        // ---- 4 B fragments + WMMA ----
        #pragma unroll
        for (int f = 0; f < 4; ++f) {
            v16bf bf;
            const unsigned short* Bp = &Bs[(wn * 64 + f * 16 + r) * G_LD];
            #pragma unroll
            for (int e = 0; e < 8; ++e) {
                bf[e]     = __builtin_bit_cast(__bf16, Bp[half * 8 + e]);
                bf[e + 8] = __builtin_bit_cast(__bf16, Bp[16 + half * 8 + e]);
            }
            acc[f] = __builtin_amdgcn_wmma_f32_16x16x32_bf16(
                false, af, false, bf, (short)0, acc[f], false, false);
        }
        wait_async0();      // DMA for tile kt+1 complete (overlapped with WMMA)
        __syncthreads();
    }
    // ---- epilogue: bias + optional SiLU, f32 and/or bf16 stores ----
    #pragma unroll
    for (int f = 0; f < 4; ++f) {
        int n = n0 + wn * 64 + f * 16 + r;
        float bv = bias[n];
        #pragma unroll
        for (int v = 0; v < 8; ++v) {
            int m = m0 + wm * 16 + half * 8 + v;
            float val = acc[f][v] + bv;
            if (act) val = silu(val);
            if (outF) outF[(size_t)m * N + n] = val;
            if (outB) outB[(size_t)m * N + n] = f2bf(val);
        }
    }
}

// ---------------------------------------------------------------------------
// Attention weights + normalized aggregation + self path.
// One block per (b,i). Single float2-vectorized pass over e_value.
// ---------------------------------------------------------------------------
__global__ __launch_bounds__(256) void attn_agg(
    const float* __restrict__ qf, const float* __restrict__ kf,
    const float* __restrict__ selfv, const float* __restrict__ e_value,
    const float* __restrict__ emask,
    float* __restrict__ v2, unsigned short* __restrict__ v2b) {
    __shared__ float qs[512];
    __shared__ float as_[128 * 32];
    __shared__ float den[32];
    const int row = blockIdx.x;            // b*128 + i
    const int b = row >> 7, i = row & 127;
    const int t = threadIdx.x;

    qs[t]       = qf[(size_t)row * 512 + t];
    qs[t + 256] = qf[(size_t)row * 512 + t + 256];
    __syncthreads();

    const float inv = 0.14433756729740643f;  // 1/sqrt(HEADS*3)
    for (int it = 0; it < 16; ++it) {
        int idx = it * 256 + t;
        int j = idx >> 5, g = idx & 31;
        const float* kr = &kf[((size_t)(b * 128 + j)) * 512 + g * 16];
        float d = 0.f;
        #pragma unroll
        for (int h = 0; h < 16; ++h) d += qs[g * 16 + h] * kr[h];
        float a = (1.f / (1.f + __expf(-d * inv))) * emask[(size_t)row * 128 + j];
        as_[j * 32 + g] = a;
    }
    __syncthreads();
    if (t < 32) {
        float s = 0.f;
        for (int j = 0; j < 128; ++j) s += as_[j * 32 + t];
        den[t] = 1.f / (s + 1e-6f);
    }
    __syncthreads();
    for (int it = 0; it < 16; ++it) {
        int idx = it * 256 + t;
        as_[idx] *= den[idx & 31];
    }
    __syncthreads();

    const int c = 2 * t;                   // adjacent channel pair, same group
    float a0 = 0.f, a1 = 0.f;
    const float* ev = &e_value[((size_t)(b * 16384 + i * 128)) * 512];
    for (int j = 0; j < 128; ++j) {
        float wgt = as_[j * 32 + (c >> 4)];
        float2 e2 = *(const float2*)&ev[(size_t)j * 512 + c];
        a0 += wgt * e2.x;
        a1 += wgt * e2.y;
    }
    float2 sv = *(const float2*)&selfv[(size_t)row * 512 + c];
    float s0 = a0 + sv.x, s1 = a1 + sv.y;
    *(float2*)&v2[(size_t)row * 512 + c] = make_float2(s0, s1);
    *(unsigned*)&v2b[(size_t)row * 512 + c] =
        (unsigned)f2bf(s0) | ((unsigned)f2bf(s1) << 16);
}

// ---------------------------------------------------------------------------
// Masked max-pool over nodes. One block per batch, float2 channel pairs.
// ---------------------------------------------------------------------------
__global__ void pool_max(const float* __restrict__ v2, const float* __restrict__ vmask,
                         unsigned short* __restrict__ poolb) {
    int b = blockIdx.x, c = 2 * threadIdx.x;
    float b0 = -3.4e38f, b1 = -3.4e38f;
    for (int k = 0; k < 128; ++k) {
        float pen = 1e9f * (1.f - vmask[b * 128 + k]);
        float2 v = *(const float2*)&v2[((size_t)(b * 128 + k)) * 512 + c];
        b0 = fmaxf(b0, v.x - pen);
        b1 = fmaxf(b1, v.y - pen);
    }
    *(unsigned*)&poolb[b * 512 + c] = (unsigned)f2bf(b0) | ((unsigned)f2bf(b1) << 16);
}

// ---------------------------------------------------------------------------
// Host-side orchestration
// ---------------------------------------------------------------------------
extern "C" void kernel_launch(void* const* d_in, const int* in_sizes, int n_in,
                              void* d_out, int out_size, void* d_ws, size_t ws_size,
                              hipStream_t stream) {
    (void)in_sizes; (void)n_in; (void)out_size; (void)ws_size;
    const float* v_f    = (const float*)d_in[0];
    const float* e_f    = (const float*)d_in[1];
    const float* p_pe   = (const float*)d_in[2];
    const float* t_pe   = (const float*)d_in[3];
    const float* emask  = (const float*)d_in[4];
    const float* vmask  = (const float*)d_in[5];
    const float* W_vpe  = (const float*)d_in[6];  const float* b_vpe  = (const float*)d_in[7];
    const float* W_epe  = (const float*)d_in[8];  const float* b_epe  = (const float*)d_in[9];
    const float* W_ev1  = (const float*)d_in[10]; const float* b_ev1  = (const float*)d_in[11];
    const float* W_ev2  = (const float*)d_in[12]; const float* b_ev2  = (const float*)d_in[13];
    const float* W_q    = (const float*)d_in[14]; const float* b_q    = (const float*)d_in[15];
    const float* W_k    = (const float*)d_in[16]; const float* b_k    = (const float*)d_in[17];
    const float* W_self = (const float*)d_in[18]; const float* b_self = (const float*)d_in[19];
    const float* W_out  = (const float*)d_in[20]; const float* b_out  = (const float*)d_in[21];

    float* v_out   = (float*)d_out;                 // (4,128,256)
    float* e_value = (float*)d_out + 131072;        // (4,16384,512)

    // ---- workspace carve (256B-aligned) ----
    char* p = (char*)d_ws;
    auto alloc = [&](size_t bytes) -> char* {
        char* r = p; p += (bytes + 255) & ~(size_t)255; return r;
    };
    unsigned short* Wvpe_p  = (unsigned short*)alloc((size_t)256 * 576 * 2);
    unsigned short* Wepe_p  = (unsigned short*)alloc((size_t)512 * 640 * 2);
    unsigned short* Wev1_p  = (unsigned short*)alloc((size_t)512 * 1024 * 2);
    unsigned short* Wev2_p  = (unsigned short*)alloc((size_t)512 * 512 * 2);
    unsigned short* Wq_p    = (unsigned short*)alloc((size_t)512 * 256 * 2);
    unsigned short* Wk_p    = (unsigned short*)alloc((size_t)512 * 256 * 2);
    unsigned short* Wself_p = (unsigned short*)alloc((size_t)512 * 256 * 2);
    unsigned short* Wout_p  = (unsigned short*)alloc((size_t)256 * 1024 * 2);
    unsigned short* vb      = (unsigned short*)alloc((size_t)512 * 256 * 2);
    unsigned short* eb      = (unsigned short*)alloc((size_t)65536 * 512 * 2);
    unsigned short* hb      = (unsigned short*)alloc((size_t)65536 * 512 * 2);
    float*          qf      = (float*)alloc((size_t)512 * 512 * 4);
    float*          kfb     = (float*)alloc((size_t)512 * 512 * 4);
    float*          selfv   = (float*)alloc((size_t)512 * 512 * 4);
    float*          v2f     = (float*)alloc((size_t)512 * 512 * 4);
    unsigned short* v2b     = (unsigned short*)alloc((size_t)512 * 512 * 2);
    unsigned short* poolb   = (unsigned short*)alloc((size_t)4 * 512 * 2);

    auto packgrid = [](int N, int Kpad) { return dim3((N * Kpad + 255) / 256); };
    pack_w<<<packgrid(256, 576),  256, 0, stream>>>(W_vpe,  Wvpe_p,  556,  256, 576);
    pack_w<<<packgrid(512, 640),  256, 0, stream>>>(W_epe,  Wepe_p,  612,  512, 640);
    pack_w<<<packgrid(512, 1024), 256, 0, stream>>>(W_ev1,  Wev1_p,  1024, 512, 1024);
    pack_w<<<packgrid(512, 512),  256, 0, stream>>>(W_ev2,  Wev2_p,  512,  512, 512);
    pack_w<<<packgrid(512, 256),  256, 0, stream>>>(W_q,    Wq_p,    256,  512, 256);
    pack_w<<<packgrid(512, 256),  256, 0, stream>>>(W_k,    Wk_p,    256,  512, 256);
    pack_w<<<packgrid(512, 256),  256, 0, stream>>>(W_self, Wself_p, 256,  512, 256);
    pack_w<<<packgrid(256, 1024), 256, 0, stream>>>(W_out,  Wout_p,  1024, 256, 1024);

    // v = silu([v_f|p_pe|t_pe] @ W_vpe + b)  -> vb (bf16)
    gemm_wmma<0><<<dim3(256 / 128, 512 / 64), 256, 0, stream>>>(
        v_f, p_pe, t_pe, nullptr, nullptr, Wvpe_p, b_vpe,
        nullptr, vb, 512, 256, 576, 1);
    // e = silu([e_f|t_pe] @ W_epe + b)  -> eb (bf16)
    gemm_wmma<1><<<dim3(512 / 128, 65536 / 64), 256, 0, stream>>>(
        e_f, nullptr, t_pe, nullptr, nullptr, Wepe_p, b_epe,
        nullptr, eb, 65536, 512, 640, 1);
    // q, k, silu(self)
    gemm_wmma<4><<<dim3(512 / 128, 512 / 64), 256, 0, stream>>>(
        nullptr, nullptr, nullptr, vb, nullptr, Wq_p, b_q,
        qf, nullptr, 512, 512, 256, 0);
    gemm_wmma<4><<<dim3(512 / 128, 512 / 64), 256, 0, stream>>>(
        nullptr, nullptr, nullptr, vb, nullptr, Wk_p, b_k,
        kfb, nullptr, 512, 512, 256, 0);
    gemm_wmma<4><<<dim3(512 / 128, 512 / 64), 256, 0, stream>>>(
        nullptr, nullptr, nullptr, vb, nullptr, Wself_p, b_self,
        selfv, nullptr, 512, 512, 256, 1);
    // h = silu([v_i|v_j|e] @ W_ev1 + b) -> hb (bf16)
    gemm_wmma<2><<<dim3(512 / 128, 65536 / 64), 256, 0, stream>>>(
        nullptr, nullptr, nullptr, vb, eb, Wev1_p, b_ev1,
        nullptr, hb, 65536, 512, 1024, 1);
    // e_value = h @ W_ev2 + b -> d_out (f32)
    gemm_wmma<3><<<dim3(512 / 128, 65536 / 64), 256, 0, stream>>>(
        nullptr, nullptr, nullptr, hb, nullptr, Wev2_p, b_ev2,
        e_value, nullptr, 65536, 512, 512, 0);
    // attention + aggregation + self  -> v2
    attn_agg<<<512, 256, 0, stream>>>(qf, kfb, selfv, e_value, emask, v2f, v2b);
    // masked max-pool
    pool_max<<<4, 256, 0, stream>>>(v2f, vmask, poolb);
    // v_out = silu([v2|pooled] @ W_out + b) -> d_out (f32)
    gemm_wmma<5><<<dim3(256 / 128, 512 / 64), 256, 0, stream>>>(
        nullptr, nullptr, nullptr, v2b, poolb, Wout_p, b_out,
        v_out, nullptr, 512, 256, 1024, 1);
}
